// FACMACDiscreteCriticGNN_8383776162077
// MI455X (gfx1250) — compile-verified, matrix operand order checked by reading, not run on hardware
//
#include <hip/hip_runtime.h>
#include <math.h>

// ---------------------------------------------------------------------------
// FACMAC discrete-critic GNN forward for MI455X (gfx1250, wave32).
//  - Node projections via v_wmma_f32_16x16x32_f16: activations pre-converted
//    to zero-padded f16 (A-fragments = two contiguous b128 loads), weights
//    pre-packed fragment-major f16 (B-fragments = two b128 loads). Each wave
//    computes a 32x128 output tile (2 row tiles, B reuse), K loop fully
//    unrolled via template, __launch_bounds__(256,1) to avoid acc spills.
//  - Edge attention: wave-per-edge, lane-parallel over 128 channels, 3-pass
//    segment softmax with L2-resident float atomics (q/k/v = 25.6MB each,
//    fit the 192MB L2; edge gathers never reach HBM).
//  - Graph LayerNorm: segment stats over node_batch with atomics.
// Param pointer decode assumes jax pytree flatten order (sorted dict keys).
// ---------------------------------------------------------------------------

typedef __attribute__((ext_vector_type(16))) _Float16 v16h;
typedef __attribute__((ext_vector_type(8)))  _Float16 v8h;
typedef __attribute__((ext_vector_type(8)))  float    v8f;

#define HID 128
#define NGRAPH 512

__device__ __forceinline__ float waveReduceSum(float v) {
#pragma unroll
  for (int off = 16; off > 0; off >>= 1) v += __shfl_xor(v, off, 32);
  return v;
}

__device__ __forceinline__ void atomicMaxF(float* addr, float val) {
  int* ia = (int*)addr;
  int old = __float_as_int(*addr);
  while (__int_as_float(old) < val) {
    int assumed = old;
    old = atomicCAS(ia, assumed, __float_as_int(val));
    if (old == assumed) break;
  }
}

__device__ __forceinline__ v16h cat16(v8h lo, v8h hi) {
  return __builtin_shufflevector(lo, hi, 0, 1, 2, 3, 4, 5, 6, 7,
                                 8, 9, 10, 11, 12, 13, 14, 15);
}

// ---------------------------------------------------------------------------
// X[n, cin] f32 -> Xh[n, cinp] f16, zero-padded columns (cinp multiple of 32)
// ---------------------------------------------------------------------------
__global__ void convert_x_f16(const float* __restrict__ X, _Float16* __restrict__ Xh,
                              int n, int cin, int cinp) {
  int i = blockIdx.x * blockDim.x + threadIdx.x;
  if (i >= n * cinp) return;
  int c = i % cinp;
  int r = i / cinp;
  Xh[i] = (c < cin) ? (_Float16)X[(size_t)r * cin + c] : (_Float16)0.0f;
}

// ---------------------------------------------------------------------------
// Pack W[cin, 128] f32 into fragment-major f16: Wp[((kk*8+nt)*32+lane)*16+el]
// B-fragment layout (32x16 f16): lane L holds col N = nt*16 + (L&15);
// elems 0..7 -> K = kk*32 + (L>>4)*8 + i, elems 8..15 -> +16.
// ---------------------------------------------------------------------------
__global__ void pack_w_f16(const float* __restrict__ W, _Float16* __restrict__ Wp,
                           int cin, int ksteps) {
  int i = blockIdx.x * blockDim.x + threadIdx.x;
  if (i >= ksteps * 4096) return;  // 8 nt * 32 lanes * 16 el = 4096 per kstep
  int el   = i & 15;
  int lane = (i >> 4) & 31;
  int nt   = (i >> 9) & 7;
  int kk   = i >> 12;
  int half = lane >> 4;
  int k = kk * 32 + ((el < 8) ? (half * 8 + el) : (16 + half * 8 + (el - 8)));
  int n = nt * 16 + (lane & 15);
  Wp[i] = (k < cin) ? (_Float16)W[(size_t)k * HID + n] : (_Float16)0.0f;
}

// ---------------------------------------------------------------------------
// WMMA GEMM: Out[nrows, 128] (+)= Xh[nrows, 32*KSTEPS] @ W + bias
// One wave handles 2 row tiles (32 rows) x 128 cols; B fragments reused.
// ---------------------------------------------------------------------------
template <int KSTEPS>
__global__ __launch_bounds__(256, 1)
void gemm_proj_f16(const _Float16* __restrict__ Xh, const _Float16* __restrict__ Wp,
                   const float* __restrict__ bias, float* __restrict__ Out,
                   int ntiles, int accumulate) {
  const int cinp = 32 * KSTEPS;
  int job = blockIdx.x * (blockDim.x >> 5) + (threadIdx.x >> 5);
  int t0 = job * 2;
  if (t0 >= ntiles) return;
  int  t1   = t0 + 1;
  bool has2 = (t1 < ntiles);
  int  t1c  = has2 ? t1 : t0;

  int lane = threadIdx.x & 31;
  int lm   = lane & 15;
  int half = lane >> 4;

  const _Float16* xr0 = Xh + (size_t)(t0 * 16 + lm) * cinp;
  const _Float16* xr1 = Xh + (size_t)(t1c * 16 + lm) * cinp;

  v8f acc0[8], acc1[8];
#pragma unroll
  for (int nt = 0; nt < 8; ++nt)
#pragma unroll
    for (int j = 0; j < 8; ++j) { acc0[nt][j] = 0.0f; acc1[nt][j] = 0.0f; }

#pragma unroll
  for (int kk = 0; kk < KSTEPS; ++kk) {
    int kb = kk * 32 + half * 8;
    v16h A0 = cat16(*(const v8h*)(xr0 + kb), *(const v8h*)(xr0 + kb + 16));
    v16h A1 = cat16(*(const v8h*)(xr1 + kb), *(const v8h*)(xr1 + kb + 16));
    const _Float16* wbase = Wp + ((size_t)kk * 8 * 32 + lane) * 16;
#pragma unroll
    for (int nt = 0; nt < 8; ++nt) {
      const _Float16* wp = wbase + (size_t)nt * 32 * 16;
      v16h B = cat16(*(const v8h*)wp, *(const v8h*)(wp + 8));
      acc0[nt] = __builtin_amdgcn_wmma_f32_16x16x32_f16(
          false, A0, false, B, (short)0, acc0[nt], false, false);
      acc1[nt] = __builtin_amdgcn_wmma_f32_16x16x32_f16(
          false, A1, false, B, (short)0, acc1[nt], false, false);
    }
  }

  if (accumulate) {
#pragma unroll
    for (int nt = 0; nt < 8; ++nt) {
      int col = nt * 16 + lm;
      float bc = bias[col];
#pragma unroll
      for (int j = 0; j < 8; ++j) {
        float* p0 = Out + (size_t)(t0 * 16 + half * 8 + j) * HID + col;
        *p0 += acc0[nt][j] + bc;
        if (has2) {
          float* p1 = Out + (size_t)(t1 * 16 + half * 8 + j) * HID + col;
          *p1 += acc1[nt][j] + bc;
        }
      }
    }
  } else {
#pragma unroll
    for (int nt = 0; nt < 8; ++nt) {
      int col = nt * 16 + lm;
      float bc = bias[col];
#pragma unroll
      for (int j = 0; j < 8; ++j) {
        Out[(size_t)(t0 * 16 + half * 8 + j) * HID + col] = acc0[nt][j] + bc;
        if (has2)
          Out[(size_t)(t1 * 16 + half * 8 + j) * HID + col] = acc1[nt][j] + bc;
      }
    }
  }
}

// --------------------------- edge attention --------------------------------
__global__ void init_softmax_bufs(float* __restrict__ amax, float* __restrict__ denom, int n2) {
  int i = blockIdx.x * blockDim.x + threadIdx.x;
  if (i < n2) { amax[i] = -3.0e38f; denom[i] = 0.0f; }
}

__global__ void edge_logits(const int* __restrict__ ei, const float* __restrict__ q,
                            const float* __restrict__ k, float* __restrict__ alpha,
                            float* __restrict__ amax, int E) {
  int e = blockIdx.x * (blockDim.x >> 5) + (threadIdx.x >> 5);
  if (e >= E) return;
  int lane = threadIdx.x & 31;
  int src = ei[e], dst = ei[E + e];
  const float* qd = q + (size_t)dst * HID;
  const float* ks = k + (size_t)src * HID;
  float s0 = qd[lane] * ks[lane] + qd[lane + 32] * ks[lane + 32];            // head 0
  float s1 = qd[lane + 64] * ks[lane + 64] + qd[lane + 96] * ks[lane + 96];  // head 1
  s0 = waveReduceSum(s0);
  s1 = waveReduceSum(s1);
  if (lane == 0) {
    s0 *= 0.125f;  // 1/sqrt(64)
    s1 *= 0.125f;
    alpha[2 * (size_t)e] = s0;
    alpha[2 * (size_t)e + 1] = s1;
    atomicMaxF(&amax[2 * dst], s0);
    atomicMaxF(&amax[2 * dst + 1], s1);
  }
}

__global__ void edge_exp(const int* __restrict__ ei, float* __restrict__ alpha,
                         const float* __restrict__ amax, float* __restrict__ denom, int E) {
  int e = blockIdx.x * blockDim.x + threadIdx.x;
  if (e >= E) return;
  int dst = ei[E + e];
  float e0 = __expf(alpha[2 * (size_t)e]     - amax[2 * dst]);
  float e1 = __expf(alpha[2 * (size_t)e + 1] - amax[2 * dst + 1]);
  alpha[2 * (size_t)e] = e0;
  alpha[2 * (size_t)e + 1] = e1;
  atomicAdd(&denom[2 * dst], e0);
  atomicAdd(&denom[2 * dst + 1], e1);
}

__global__ void edge_agg(const int* __restrict__ ei, const float* __restrict__ ex,
                         const float* __restrict__ denom, const float* __restrict__ v,
                         float* __restrict__ out, int E) {
  int e = blockIdx.x * (blockDim.x >> 5) + (threadIdx.x >> 5);
  if (e >= E) return;
  int lane = threadIdx.x & 31;
  int src = ei[e], dst = ei[E + e];
  float a0 = ex[2 * (size_t)e]     / fmaxf(denom[2 * dst], 1e-16f);
  float a1 = ex[2 * (size_t)e + 1] / fmaxf(denom[2 * dst + 1], 1e-16f);
  const float* vs = v + (size_t)src * HID;
  float* od = out + (size_t)dst * HID;
  atomicAdd(&od[lane],      vs[lane]      * a0);
  atomicAdd(&od[lane + 32], vs[lane + 32] * a0);
  atomicAdd(&od[lane + 64], vs[lane + 64] * a1);
  atomicAdd(&od[lane + 96], vs[lane + 96] * a1);
}

// ---------------------------- graph LayerNorm ------------------------------
__global__ void zero_buf(float* __restrict__ p, int n) {
  int i = blockIdx.x * blockDim.x + threadIdx.x;
  if (i < n) p[i] = 0.0f;
}

__global__ void ln_stats_sum(const float* __restrict__ hn, const int* __restrict__ batch,
                             float* __restrict__ gsum, float* __restrict__ gcnt, int n) {
  int node = blockIdx.x * (blockDim.x >> 5) + (threadIdx.x >> 5);
  if (node >= n) return;
  int lane = threadIdx.x & 31;
  const float* r = hn + (size_t)node * HID;
  float s = fmaxf(r[lane], 0.f) + fmaxf(r[lane + 32], 0.f) +
            fmaxf(r[lane + 64], 0.f) + fmaxf(r[lane + 96], 0.f);
  s = waveReduceSum(s);
  if (lane == 0) {
    int g = batch[node];
    atomicAdd(&gsum[g], s);
    atomicAdd(&gcnt[g], 1.0f);
  }
}

__global__ void ln_mean(const float* __restrict__ gsum, const float* __restrict__ gcnt,
                        float* __restrict__ gmean, int g) {
  int i = blockIdx.x * blockDim.x + threadIdx.x;
  if (i < g) {
    float c = fmaxf(gcnt[i] * (float)HID, 1.0f);
    gmean[i] = gsum[i] / c;
  }
}

__global__ void ln_stats_var(const float* __restrict__ hn, const int* __restrict__ batch,
                             const float* __restrict__ gmean, float* __restrict__ gvar, int n) {
  int node = blockIdx.x * (blockDim.x >> 5) + (threadIdx.x >> 5);
  if (node >= n) return;
  int lane = threadIdx.x & 31;
  int g = batch[node];
  float m = gmean[g];
  const float* r = hn + (size_t)node * HID;
  float s = 0.f;
#pragma unroll
  for (int j = 0; j < 4; ++j) {
    float xc = fmaxf(r[lane + 32 * j], 0.f) - m;
    s += xc * xc;
  }
  s = waveReduceSum(s);
  if (lane == 0) atomicAdd(&gvar[g], s);
}

__global__ void ln_rstd(const float* __restrict__ gvar, const float* __restrict__ gcnt,
                        float* __restrict__ grstd, int g) {
  int i = blockIdx.x * blockDim.x + threadIdx.x;
  if (i < g) {
    float c = fmaxf(gcnt[i] * (float)HID, 1.0f);
    grstd[i] = rsqrtf(gvar[i] / c + 1e-5f);
  }
}

__global__ void ln_apply(const float* __restrict__ hn, const int* __restrict__ batch,
                         const float* __restrict__ gmean, const float* __restrict__ grstd,
                         const float* __restrict__ lw, const float* __restrict__ lb,
                         float* __restrict__ hout, int n) {
  int node = blockIdx.x;
  int f = threadIdx.x;
  if (node >= n) return;
  int g = batch[node];
  float val = fmaxf(hn[(size_t)node * HID + f], 0.f);
  hout[(size_t)node * HID + f] = (val - gmean[g]) * grstd[g] * lw[f] + lb[f];
}

// ------------------------------ output head --------------------------------
__global__ void out_proj(const float* __restrict__ h, const float* __restrict__ Wout,
                         const float* __restrict__ bout, float* __restrict__ out, int n) {
  int i = blockIdx.x * blockDim.x + threadIdx.x;
  if (i >= n) return;
  float s = bout[0];
  const float* r = h + (size_t)i * HID;
#pragma unroll 8
  for (int f = 0; f < HID; ++f) s += r[f] * Wout[f];
  out[i] = s;
}

// ---------------------------------------------------------------------------
extern "C" void kernel_launch(void* const* d_in, const int* in_sizes, int n_in,
                              void* d_out, int out_size, void* d_ws, size_t ws_size,
                              hipStream_t stream) {
  const float* x      = (const float*)d_in[0];
  const int*   ei_ue  = (const int*)d_in[1];
  const int*   ei_ap  = (const int*)d_in[2];
  const int*   nbatch = (const int*)d_in[3];
  const float* Wout   = (const float*)d_in[4];
  const float* bout   = (const float*)d_in[5];

  const int N    = in_sizes[3];
  const int E    = in_sizes[1] / 2;
  const int F_IN = in_sizes[0] / N;

  // pytree flatten (sorted keys) per layer:
  // ap{Wk,Wq,Wskip,Wv,bk,bq,bskip,bv}, ln_b, ln_w, ue{Wk,Wq,Wskip,Wv,bk,bq,bskip,bv}
  struct ConvP { const float *Wq, *bq, *Wk, *bk, *Wv, *bv, *Ws, *bs; };
  struct LayerP { ConvP ue, ap; const float *lnw, *lnb; };
  LayerP L[2];
  int idx = 6;
  for (int l = 0; l < 2; ++l) {
    ConvP ap;
    ap.Wk = (const float*)d_in[idx++]; ap.Wq = (const float*)d_in[idx++];
    ap.Ws = (const float*)d_in[idx++]; ap.Wv = (const float*)d_in[idx++];
    ap.bk = (const float*)d_in[idx++]; ap.bq = (const float*)d_in[idx++];
    ap.bs = (const float*)d_in[idx++]; ap.bv = (const float*)d_in[idx++];
    const float* lnb = (const float*)d_in[idx++];
    const float* lnw = (const float*)d_in[idx++];
    ConvP ue;
    ue.Wk = (const float*)d_in[idx++]; ue.Wq = (const float*)d_in[idx++];
    ue.Ws = (const float*)d_in[idx++]; ue.Wv = (const float*)d_in[idx++];
    ue.bk = (const float*)d_in[idx++]; ue.bq = (const float*)d_in[idx++];
    ue.bs = (const float*)d_in[idx++]; ue.bv = (const float*)d_in[idx++];
    L[l].ue = ue; L[l].ap = ap; L[l].lnw = lnw; L[l].lnb = lnb;
  }

  // workspace carve (~150 MB)
  float* ws = (float*)d_ws;
  float* h     = ws; ws += (size_t)N * HID;
  float* hn    = ws; ws += (size_t)N * HID;
  float* qb    = ws; ws += (size_t)N * HID;
  float* kb    = ws; ws += (size_t)N * HID;
  float* vb    = ws; ws += (size_t)N * HID;
  float* alpha = ws; ws += (size_t)E * 2;
  float* amax  = ws; ws += (size_t)N * 2;
  float* denom = ws; ws += (size_t)N * 2;
  float* gsum  = ws; ws += NGRAPH;   // gsum/gcnt/gvar contiguous for one zero pass
  float* gcnt  = ws; ws += NGRAPH;
  float* gvar  = ws; ws += NGRAPH;
  float* gmean = ws; ws += NGRAPH;
  float* grstd = ws; ws += NGRAPH;
  _Float16* Xh = (_Float16*)ws; ws += (size_t)N * HID / 2;   // f16 activations
  _Float16* Wp = (_Float16*)ws; ws += (4 * 4096) / 2;        // packed weights (max 4 ksteps)

  const int ntiles = N / 16;
  const int jobs = (ntiles + 1) / 2;
  dim3 gemmGrid((jobs + 7) / 8), gemmBlk(256);

  for (int l = 0; l < 2; ++l) {
    const float* inX = (l == 0) ? x : h;
    int cin  = (l == 0) ? F_IN : HID;
    int cinp = (cin + 31) & ~31;     // 64 for layer0, 128 for layer1
    int ksteps = cinp / 32;

    convert_x_f16<<<(N * cinp + 255) / 256, 256, 0, stream>>>(inX, Xh, N, cin, cinp);

    auto gemm = [&](const float* W, const float* b, float* O, int acc) {
      pack_w_f16<<<(ksteps * 4096 + 255) / 256, 256, 0, stream>>>(W, Wp, cin, ksteps);
      if (ksteps == 2)
        gemm_proj_f16<2><<<gemmGrid, gemmBlk, 0, stream>>>(Xh, Wp, b, O, ntiles, acc);
      else
        gemm_proj_f16<4><<<gemmGrid, gemmBlk, 0, stream>>>(Xh, Wp, b, O, ntiles, acc);
    };

    // root-weight (skip) projections, summed over both relations, into hn
    gemm(L[l].ue.Ws, L[l].ue.bs, hn, 0);
    gemm(L[l].ap.Ws, L[l].ap.bs, hn, 1);

    const int* eis[2] = {ei_ue, ei_ap};
    const ConvP* cps[2] = {&L[l].ue, &L[l].ap};
    for (int r = 0; r < 2; ++r) {
      const ConvP& c = *cps[r];
      const int* ei = eis[r];
      gemm(c.Wq, c.bq, qb, 0);
      gemm(c.Wk, c.bk, kb, 0);
      gemm(c.Wv, c.bv, vb, 0);
      init_softmax_bufs<<<(2 * N + 255) / 256, 256, 0, stream>>>(amax, denom, 2 * N);
      edge_logits<<<(E + 7) / 8, 256, 0, stream>>>(ei, qb, kb, alpha, amax, E);
      edge_exp<<<(E + 255) / 256, 256, 0, stream>>>(ei, alpha, amax, denom, E);
      edge_agg<<<(E + 7) / 8, 256, 0, stream>>>(ei, alpha, denom, vb, hn, E);
    }

    // relu + graph LayerNorm -> h
    zero_buf<<<(3 * NGRAPH + 255) / 256, 256, 0, stream>>>(gsum, 3 * NGRAPH);
    ln_stats_sum<<<(N + 7) / 8, 256, 0, stream>>>(hn, nbatch, gsum, gcnt, N);
    ln_mean<<<(NGRAPH + 255) / 256, 256, 0, stream>>>(gsum, gcnt, gmean, NGRAPH);
    ln_stats_var<<<(N + 7) / 8, 256, 0, stream>>>(hn, nbatch, gmean, gvar, N);
    ln_rstd<<<(NGRAPH + 255) / 256, 256, 0, stream>>>(gvar, gcnt, grstd, NGRAPH);
    ln_apply<<<N, HID, 0, stream>>>(hn, nbatch, gmean, grstd, L[l].lnw, L[l].lnb, h, N);
  }

  out_proj<<<(N + 255) / 256, 256, 0, stream>>>(h, Wout, bout, (float*)d_out, N);
}
